// Model2D_34273839022223
// MI455X (gfx1250) — matrix operand order repaired and checked by reading, not compiled, not run on hardware
//
#include <hip/hip_runtime.h>
#include <hip/hip_bf16.h>

typedef __attribute__((ext_vector_type(16))) __bf16 v16bf;
typedef __attribute__((ext_vector_type(8)))  __bf16 v8bf;
typedef __attribute__((ext_vector_type(8)))  float  v8f;

#define NEG_ATT 0.2f
#define NEG_RES 0.01f

// ---------- helpers: order-preserving float<->uint key for atomicMax ----------
__device__ __forceinline__ unsigned float_to_key(float f) {
    unsigned b = __float_as_uint(f);
    return (b & 0x80000000u) ? ~b : (b | 0x80000000u);
}
__device__ __forceinline__ float key_to_float(unsigned key) {
    unsigned b = (key & 0x80000000u) ? (key ^ 0x80000000u) : ~key;
    return __uint_as_float(b);
}

// ---------- build h0 = concat(x, signal), bf16, stride 96, zero padded ----------
__global__ void build_h0(const float* __restrict__ x, const float* __restrict__ sig,
                         __bf16* __restrict__ hpad, int N) {
    int idx = blockIdx.x * blockDim.x + threadIdx.x;
    int n = idx / 96, c = idx % 96;
    if (n >= N) return;
    float v = (c < 64) ? x[(size_t)n * 64 + c] : (c == 64 ? sig[n] : 0.0f);
    hpad[(size_t)n * 96 + c] = (__bf16)v;
}

// ---------- pack weight matrix [d_in,64] fp32 -> WMMA B-fragment layout, bf16 ----------
// layout: frag[((kt*4+ct)*32 + lane)*16 + e]; (k,n) = (kt*32 + (lane>>4)*16 + e, ct*16 + (lane&15))
__global__ void pack_w(const float* __restrict__ W, __bf16* __restrict__ out,
                       int d_in, int total) {
    int idx = blockIdx.x * blockDim.x + threadIdx.x;
    if (idx >= total) return;
    int e     = idx & 15;
    int lane  = (idx >> 4) & 31;
    int ct    = (idx >> 9) & 3;
    int kt    = idx >> 11;
    int n = ct * 16 + (lane & 15);
    int k = kt * 32 + (lane >> 4) * 16 + e;
    out[idx] = (k < d_in) ? (__bf16)W[k * 64 + n] : (__bf16)0.0f;
}

// ---------- node GEMM: xl = h@Wl, xr = h@Wr via WMMA bf16 ----------
// Block = 256 threads = 8 waves: waves 0-3 -> Wl (4 row tiles), waves 4-7 -> Wr.
// Each wave holds all B fragments in registers and grid-strides over 16-row tiles.
template <int NKT>
__global__ void __launch_bounds__(256) node_gemm(
    const __bf16* __restrict__ hpad,
    const __bf16* __restrict__ WpackL, const __bf16* __restrict__ WpackR,
    float* __restrict__ xl, float* __restrict__ xr, int nrows)
{
    int lane  = threadIdx.x & 31;
    int wv    = threadIdx.x >> 5;
    int mat   = wv >> 2;
    int tsub  = wv & 3;
    const __bf16* Wp = mat ? WpackR : WpackL;
    float*       out = mat ? xr : xl;
    int hi = lane >> 4, mlane = lane & 15;

    v16bf bfrag[NKT][4];
    #pragma unroll
    for (int kt = 0; kt < NKT; ++kt)
        #pragma unroll
        for (int ct = 0; ct < 4; ++ct)
            bfrag[kt][ct] = *(const v16bf*)(Wp + (((kt * 4 + ct) * 32 + lane) << 4));

    int ntiles = (nrows + 15) >> 4;
    for (int t = blockIdx.x * 4 + tsub; t < ntiles; t += gridDim.x * 4) {
        int row0 = t * 16;
        int arow = row0 + mlane;
        if (arow > nrows - 1) arow = nrows - 1;   // clamp; stores guarded on tail
        const __bf16* hrow = hpad + (size_t)arow * 96;

        v8f accv[4] = {};
        #pragma unroll
        for (int kt = 0; kt < NKT; ++kt) {
            int ka = kt * 32 + hi * 16;
            v8bf alo = *(const v8bf*)(hrow + ka);
            v8bf ahi = *(const v8bf*)(hrow + ka + 8);
            v16bf a = __builtin_shufflevector(alo, ahi,
                0, 1, 2, 3, 4, 5, 6, 7, 8, 9, 10, 11, 12, 13, 14, 15);
            #pragma unroll
            for (int ct = 0; ct < 4; ++ct)
                accv[ct] = __builtin_amdgcn_wmma_f32_16x16x32_bf16(
                    false, a, false, bfrag[kt][ct], (short)0, accv[ct], false, false);
        }

        if (row0 + 16 <= nrows) {                 // fast path: full tile
            #pragma unroll
            for (int ct = 0; ct < 4; ++ct) {
                int n = ct * 16 + mlane;
                #pragma unroll
                for (int i = 0; i < 8; ++i)
                    out[(size_t)(row0 + i + hi * 8) * 64 + n] = accv[ct][i];
            }
        } else {
            #pragma unroll
            for (int ct = 0; ct < 4; ++ct) {
                int n = ct * 16 + mlane;
                #pragma unroll
                for (int i = 0; i < 8; ++i) {
                    int m = row0 + i + hi * 8;
                    if (m < nrows) out[(size_t)m * 64 + n] = accv[ct][i];
                }
            }
        }
    }
}

// ---------- zero per-layer segment state ----------
__global__ void seg_init(float* __restrict__ acc, unsigned* __restrict__ amax,
                         float* __restrict__ denom, int N) {
    int idx = blockIdx.x * blockDim.x + threadIdx.x;
    if (idx < N * 64) acc[idx] = 0.0f;
    if (idx < N) { amax[idx] = 0u; denom[idx] = 0.0f; }
}

// ---------- pass 1: alpha[e] = leaky(m)@att ; segment max via key atomics ----------
// one wave per edge, 2 features per lane -> coalesced 128B row gathers
__global__ void __launch_bounds__(256) edge_alpha(
    const int* __restrict__ src, const int* __restrict__ dst,
    const float* __restrict__ edge_attr, const float* __restrict__ We,
    const float* __restrict__ attv, const float* __restrict__ xl,
    const float* __restrict__ xr, float* __restrict__ alpha,
    unsigned* __restrict__ amax, int E)
{
    __shared__ float sWe[640];
    __shared__ float sAtt[64];
    for (int i = threadIdx.x; i < 640; i += blockDim.x) sWe[i] = We[i];
    for (int i = threadIdx.x; i < 64;  i += blockDim.x) sAtt[i] = attv[i];
    __syncthreads();

    int lane = threadIdx.x & 31;
    int wid  = (int)((blockIdx.x * blockDim.x + threadIdx.x) >> 5);
    int nw   = (int)((gridDim.x * blockDim.x) >> 5);
    float a0 = sAtt[lane], a1 = sAtt[lane + 32];

    for (int e = wid; e < E; e += nw) {
        int s = src[e], d = dst[e];
        float eav = (lane < 10) ? edge_attr[(size_t)e * 10 + lane] : 0.0f;
        float v0 = xl[(size_t)s * 64 + lane]      + xr[(size_t)d * 64 + lane];
        float v1 = xl[(size_t)s * 64 + lane + 32] + xr[(size_t)d * 64 + lane + 32];
        #pragma unroll
        for (int k = 0; k < 10; ++k) {          // e_edge = edge_attr @ We, inline
            float ek = __shfl(eav, k, 32);
            v0 += ek * sWe[k * 64 + lane];
            v1 += ek * sWe[k * 64 + lane + 32];
        }
        float l0 = v0 > 0.0f ? v0 : NEG_ATT * v0;
        float l1 = v1 > 0.0f ? v1 : NEG_ATT * v1;
        float p  = l0 * a0 + l1 * a1;
        #pragma unroll
        for (int o = 16; o > 0; o >>= 1) p += __shfl_xor(p, o, 32);
        if (lane == 0) {
            alpha[e] = p;
            atomicMax(&amax[d], float_to_key(p));
        }
    }
}

// ---------- pass 2: ex = exp(alpha - amax[dst]); denom += ex ----------
__global__ void edge_exp(const int* __restrict__ dst, const float* __restrict__ alpha,
                         const unsigned* __restrict__ amax, float* __restrict__ exb,
                         float* __restrict__ denom, int E) {
    int e = blockIdx.x * blockDim.x + threadIdx.x;
    if (e >= E) return;
    int d = dst[e];
    float v = __expf(alpha[e] - key_to_float(amax[d]));
    exb[e] = v;
    atomicAdd(&denom[d], v);
}

// ---------- pass 3: acc[dst] += (ex/denom[dst]) * xl[src] ----------
__global__ void __launch_bounds__(256) edge_scatter(
    const int* __restrict__ src, const int* __restrict__ dst,
    const float* __restrict__ exb, const float* __restrict__ denom,
    const float* __restrict__ xl, float* __restrict__ acc, int E)
{
    int lane = threadIdx.x & 31;
    int wid  = (int)((blockIdx.x * blockDim.x + threadIdx.x) >> 5);
    int nw   = (int)((gridDim.x * blockDim.x) >> 5);
    for (int e = wid; e < E; e += nw) {
        int s = src[e], d = dst[e];
        float w = exb[e] / (denom[d] + 1e-16f);
        atomicAdd(&acc[(size_t)d * 64 + lane],      w * xl[(size_t)s * 64 + lane]);
        atomicAdd(&acc[(size_t)d * 64 + lane + 32], w * xl[(size_t)s * 64 + lane + 32]);
    }
}

// ---------- epilogue: +bias, residual combine, repack bf16 features ----------
__global__ void epilogue(const float* __restrict__ acc, const float* __restrict__ bias,
                         const float* __restrict__ hprev, float* __restrict__ hout,
                         __bf16* __restrict__ hpad, int layer, int N) {
    int idx = blockIdx.x * blockDim.x + threadIdx.x;
    if (idx >= N * 64) return;
    int n = idx >> 6, f = idx & 63;
    float v = acc[idx] + bias[f];
    if (layer) {
        float lr = v > 0.0f ? v : NEG_RES * v;
        v = 0.5f * (lr + hprev[idx]);
    }
    hout[idx] = v;
    hpad[(size_t)n * 96 + f] = (__bf16)v;
}

extern "C" void kernel_launch(void* const* d_in, const int* in_sizes, int n_in,
                              void* d_out, int out_size, void* d_ws, size_t ws_size,
                              hipStream_t stream) {
    (void)n_in; (void)out_size; (void)ws_size;
    const float* x      = (const float*)d_in[0];
    const float* signal = (const float*)d_in[1];
    const int*   eidx   = (const int*)d_in[2];
    const float* eattr  = (const float*)d_in[3];
    int N = in_sizes[1];            // signal has N elements
    int E = in_sizes[3] / 10;       // edge_attr is [E,10]
    const int* src = eidx;
    const int* dst = eidx + E;

    // params pytree flattening (dict keys sorted): We, Wl, Wr, att, bias per layer
    const float *We[3], *Wl[3], *Wr[3], *att[3], *bias[3];
    int din[3];
    for (int l = 0; l < 3; ++l) {
        We[l]   = (const float*)d_in[4 + 5 * l];
        Wl[l]   = (const float*)d_in[5 + 5 * l];
        Wr[l]   = (const float*)d_in[6 + 5 * l];
        att[l]  = (const float*)d_in[7 + 5 * l];
        bias[l] = (const float*)d_in[8 + 5 * l];
        din[l]  = in_sizes[5 + 5 * l] / 64;     // 65, 64, 64
    }

    // workspace carve (≈135 MB)
    char* ws = (char*)d_ws; size_t off = 0;
    auto carve = [&](size_t bytes) -> char* {
        char* p = ws + off; off = (off + bytes + 255) & ~(size_t)255; return p;
    };
    __bf16*  hpad  = (__bf16*)carve((size_t)N * 96 * 2);
    float*   h     = (float*)carve((size_t)N * 64 * 4);
    float*   xl    = (float*)carve((size_t)N * 64 * 4);
    float*   xr    = (float*)carve((size_t)N * 64 * 4);
    float*   alpha = (float*)carve((size_t)E * 4);
    float*   exb   = (float*)carve((size_t)E * 4);
    unsigned* amax = (unsigned*)carve((size_t)N * 4);
    float*   denom = (float*)carve((size_t)N * 4);
    float*   acc   = (float*)carve((size_t)N * 64 * 4);
    __bf16*  wpkL  = (__bf16*)carve((size_t)3 * 4 * 32 * 16 * 2);   // max NKT=3
    __bf16*  wpkR  = (__bf16*)carve((size_t)3 * 4 * 32 * 16 * 2);

    build_h0<<<(N * 96 + 255) / 256, 256, 0, stream>>>(x, signal, hpad, N);

    int ntiles  = (N + 15) / 16;
    int gblocks = (ntiles + 3) / 4;

    for (int l = 0; l < 3; ++l) {
        int nkt   = (din[l] + 31) / 32;          // 3, 2, 2
        int total = nkt * 4 * 32 * 16;
        pack_w<<<(total + 255) / 256, 256, 0, stream>>>(Wl[l], wpkL, din[l], total);
        pack_w<<<(total + 255) / 256, 256, 0, stream>>>(Wr[l], wpkR, din[l], total);
        if (nkt == 3)
            node_gemm<3><<<gblocks, 256, 0, stream>>>(hpad, wpkL, wpkR, xl, xr, N);
        else
            node_gemm<2><<<gblocks, 256, 0, stream>>>(hpad, wpkL, wpkR, xl, xr, N);
        seg_init<<<(N * 64 + 255) / 256, 256, 0, stream>>>(acc, amax, denom, N);
        edge_alpha<<<(E * 32 + 255) / 256, 256, 0, stream>>>(src, dst, eattr, We[l], att[l],
                                                             xl, xr, alpha, amax, E);
        edge_exp<<<(E + 255) / 256, 256, 0, stream>>>(dst, alpha, amax, exb, denom, E);
        edge_scatter<<<(E * 32 + 255) / 256, 256, 0, stream>>>(src, dst, exb, denom, xl, acc, E);
        float* hdst = (l == 2) ? (float*)d_out : h;
        epilogue<<<(N * 64 + 255) / 256, 256, 0, stream>>>(acc, bias[l], h, hdst, hpad, l, N);
    }
}
